// FasterRCNN_85409719648786
// MI455X (gfx1250) — compile-verified, matrix operand order checked by reading, not run on hardware
//
#include <hip/hip_runtime.h>
#include <hip/hip_bf16.h>
#include <math.h>

typedef __attribute__((ext_vector_type(16))) __bf16 v16bf;
typedef __attribute__((ext_vector_type(8)))  float  v8f;
typedef unsigned int u32x4 __attribute__((ext_vector_type(4)));
typedef int          i32x8 __attribute__((ext_vector_type(8)));
typedef int          i32x4 __attribute__((ext_vector_type(4)));

#define IMGSZ   800.0f
#define NROIS   22500
#define PRE_NMS 2000
#define POST_NMS 128
#define IOU_TH  0.7f
#define FEAT_C  512
#define FEAT_HW 50
#define D_IN    25088
#define D_H     4096
#define NC      21

__device__ __forceinline__ unsigned short f2bf_bits(float f) {
    unsigned u = __float_as_uint(f);
    u += 0x7FFFu + ((u >> 16) & 1u);          // round-to-nearest-even
    return (unsigned short)(u >> 16);
}

// ---------------------------------------------------------------------------
// 1) Clip ROIs, validity mask, softmax score (column 1)
// ---------------------------------------------------------------------------
__global__ __launch_bounds__(256) void rpn_score_clip(
        const float* __restrict__ rois, const float* __restrict__ logits,
        float* __restrict__ clipped, float* __restrict__ scores) {
    int i = blockIdx.x * 256 + threadIdx.x;
    if (i >= NROIS) return;
    float y1 = fminf(fmaxf(rois[i*4+0], 0.f), IMGSZ);
    float x1 = fminf(fmaxf(rois[i*4+1], 0.f), IMGSZ);
    float y2 = fminf(fmaxf(rois[i*4+2], 0.f), IMGSZ);
    float x2 = fminf(fmaxf(rois[i*4+3], 0.f), IMGSZ);
    clipped[i*4+0] = y1; clipped[i*4+1] = x1;
    clipped[i*4+2] = y2; clipped[i*4+3] = x2;
    bool valid = ((y2 - y1) >= 16.f) && ((x2 - x1) >= 16.f);
    float l0 = logits[i*2+0], l1 = logits[i*2+1];
    float s = 1.f / (1.f + __expf(l0 - l1));   // softmax[:,1]
    scores[i] = valid ? s : -1.f;
}

// ---------------------------------------------------------------------------
// 2) Stable top-PRE_NMS by rank counting (deterministic ordering for NMS).
// ---------------------------------------------------------------------------
__global__ __launch_bounds__(256) void rank_topk(
        const float* __restrict__ scores, int* __restrict__ order) {
    __shared__ float ks[256];
    int tid = threadIdx.x;
    int i = blockIdx.x * 256 + tid;
    float ki = (i < NROIS) ? scores[i] : -3.0f;
    int cnt = 0;
    for (int t = 0; t < NROIS; t += 256) {
        int j = t + tid;
        ks[tid] = (j < NROIS) ? scores[j] : -3.0f;
        __syncthreads();
        int lim = min(256, NROIS - t);
        for (int jj = 0; jj < lim; ++jj) {
            float kj = ks[jj];
            int jg = t + jj;
            cnt += (kj > ki || (kj == ki && jg < i)) ? 1 : 0;
        }
        int done = (cnt >= PRE_NMS) ? 1 : 0;
        if (__syncthreads_and(done)) break;    // uniform break (also the barrier)
    }
    if (i < NROIS && cnt < PRE_NMS) order[cnt] = i;
}

// ---------------------------------------------------------------------------
// 3) Greedy NMS on one wave32, then compact first POST_NMS kept boxes.
// ---------------------------------------------------------------------------
__global__ __launch_bounds__(32) void nms_compact(
        const float* __restrict__ clipped, const int* __restrict__ order,
        float* __restrict__ outRois, float* __restrict__ roisS) {
    __shared__ float by1[PRE_NMS], bx1[PRE_NMS], by2[PRE_NMS], bx2[PRE_NMS], bar[PRE_NMS];
    __shared__ int keep[PRE_NMS];
    __shared__ int sel[POST_NMS];
    int lane = threadIdx.x;
    for (int j = lane; j < PRE_NMS; j += 32) {
        int o = order[j];
        o = (o < 0) ? 0 : ((o >= NROIS) ? 0 : o);
        float y1 = clipped[o*4+0], x1 = clipped[o*4+1];
        float y2 = clipped[o*4+2], x2 = clipped[o*4+3];
        by1[j] = y1; bx1[j] = x1; by2[j] = y2; bx2[j] = x2;
        bar[j] = (y2 - y1) * (x2 - x1);
        keep[j] = 1;
    }
    __syncthreads();
    for (int i = 0; i < PRE_NMS; ++i) {
        if (keep[i]) {
            float y1 = by1[i], x1 = bx1[i], y2 = by2[i], x2 = bx2[i], ai = bar[i];
            for (int j = i + 1 + lane; j < PRE_NMS; j += 32) {
                if (!keep[j]) continue;
                float yy1 = fmaxf(y1, by1[j]), xx1 = fmaxf(x1, bx1[j]);
                float yy2 = fminf(y2, by2[j]), xx2 = fminf(x2, bx2[j]);
                float inter = fmaxf(yy2 - yy1, 0.f) * fmaxf(xx2 - xx1, 0.f);
                float iou = inter / (ai + bar[j] - inter + 1e-9f);
                if (iou > IOU_TH) keep[j] = 0;
            }
        }
        __syncthreads();
    }
    if (lane == 0) {
        int cnt = 0;
        for (int j = 0; j < PRE_NMS && cnt < POST_NMS; ++j) if (keep[j])  sel[cnt++] = j;
        for (int j = 0; j < PRE_NMS && cnt < POST_NMS; ++j) if (!keep[j]) sel[cnt++] = j;
    }
    __syncthreads();
    for (int r = lane; r < POST_NMS; r += 32) {
        int j = sel[r];
        float v0 = by1[j], v1 = bx1[j], v2 = by2[j], v3 = bx2[j];
        outRois[r*4+0] = v0; outRois[r*4+1] = v1; outRois[r*4+2] = v2; outRois[r*4+3] = v3;
        roisS[r*4+0] = v0 * 0.0625f; roisS[r*4+1] = v1 * 0.0625f;
        roisS[r*4+2] = v2 * 0.0625f; roisS[r*4+3] = v3 * 0.0625f;
    }
}

// ---------------------------------------------------------------------------
// 4) RoIAlign (7x7, 2x2 sampling, avg) -> pooled features written in bf16
// ---------------------------------------------------------------------------
__global__ __launch_bounds__(256) void roi_align_bf16(
        const float* __restrict__ feat, const float* __restrict__ roisS,
        unsigned short* __restrict__ pooled) {
    int o = blockIdx.x * 256 + threadIdx.x;      // total = 128*512*49 (exact grid)
    int r = o / (FEAT_C * 49);
    int rem = o - r * (FEAT_C * 49);
    int c = rem / 49;
    int cell = rem - c * 49;
    int oy = cell / 7, ox = cell - oy * 7;
    float y1 = roisS[r*4+0], x1 = roisS[r*4+1], y2 = roisS[r*4+2], x2 = roisS[r*4+3];
    float bh = fmaxf(y2 - y1, 1.f) * (1.f / 7.f);
    float bw = fmaxf(x2 - x1, 1.f) * (1.f / 7.f);
    const float* fc = feat + (size_t)c * (FEAT_HW * FEAT_HW);
    float acc = 0.f;
    #pragma unroll
    for (int s = 0; s < 4; ++s) {
        int sy = s >> 1, sx = s & 1;
        float gy = ((float)(oy * 2 + sy) + 0.5f) * 0.5f;
        float gx = ((float)(ox * 2 + sx) + 0.5f) * 0.5f;
        float ys = fminf(fmaxf(y1 + gy * bh, 0.f), (float)(FEAT_HW - 1));
        float xs = fminf(fmaxf(x1 + gx * bw, 0.f), (float)(FEAT_HW - 1));
        float y0f = floorf(ys), x0f = floorf(xs);
        float ly = ys - y0f, lx = xs - x0f;
        int y0 = (int)y0f, x0 = (int)x0f;
        int yb = min(y0 + 1, FEAT_HW - 1), xb = min(x0 + 1, FEAT_HW - 1);
        float v00 = fc[y0*FEAT_HW + x0], v01 = fc[y0*FEAT_HW + xb];
        float v10 = fc[yb*FEAT_HW + x0], v11 = fc[yb*FEAT_HW + xb];
        acc += v00*(1.f-ly)*(1.f-lx) + v01*(1.f-ly)*lx + v10*ly*(1.f-lx) + v11*ly*lx;
    }
    pooled[(size_t)r * D_IN + c * 49 + cell] = f2bf_bits(acc * 0.25f);
}

// ---------------------------------------------------------------------------
// TDM: issue one 2D tile load (Global -> LDS) via Tensor Data Mover.
// D# packing per CDNA5 ISA.
// tensor dims == tile dims (tile fully in-bounds), row stride = leading dim,
// LDS padding done by the DMA engine (pad_interval/pad_amount).
// word0flags carries data_size|pad_enable|pad_interval|pad_amount bits.
// ---------------------------------------------------------------------------
__device__ __forceinline__ void tdm_load_2d(
        unsigned ldsAddr, unsigned long long gaddr,
        unsigned word0flags, int tileX, int tileY, long long strideElems) {
    u32x4 g0;
    g0[0] = 1u;                                   // count=1, user mode
    g0[1] = ldsAddr;                              // lds_addr (bytes)
    g0[2] = (unsigned)(gaddr & 0xffffffffull);    // global_addr[31:0]
    g0[3] = (unsigned)((gaddr >> 32) & 0x01ffffffull) | (2u << 30); // [56:32] | type=2
    i32x8 g1;
    g1[0] = (int)word0flags;                      // mask=0 | data_size | pad ctl
    g1[1] = (int)(((unsigned)tileX & 0xffffu) << 16);               // tensor_dim0 lo
    g1[2] = (int)((((unsigned)tileX) >> 16) | (((unsigned)tileY & 0xffffu) << 16));
    g1[3] = (int)((((unsigned)tileY) >> 16) | (((unsigned)tileX) << 16)); // | tile_dim0
    g1[4] = (int)(unsigned)tileY;                 // tile_dim1 (tile_dim2 = 0)
    g1[5] = (int)(unsigned)(strideElems & 0xffffffffll);            // dim0_stride lo
    g1[6] = (int)(unsigned)((strideElems >> 32) & 0xffffll);        // dim0_stride hi
    g1[7] = 0;
    i32x4 z4 = {0, 0, 0, 0};
#if __clang_major__ >= 23
    i32x8 z8 = {0, 0, 0, 0, 0, 0, 0, 0};
    __builtin_amdgcn_tensor_load_to_lds(g0, g1, z4, z4, z8, 0);
#else
    __builtin_amdgcn_tensor_load_to_lds(g0, g1, z4, z4, 0);
#endif
}

// ---------------------------------------------------------------------------
// 5) GEMM: C(128,N) = A_bf16(128,K) @ W_f32(K,N) + bias ; bf16 WMMA, f32 acc.
//    Aligned path: double-buffered TDM staging of both tiles (TENSORcnt
//    ping-pong, s_wait_tensorcnt(2) while next tile is in flight).
//    A tile 128x32 bf16 -> LDS rows padded to 80B (pad 4 dw / 16 dw).
//    W tile  32x128 f32 -> LDS rows padded to 132 floats (pad 4 dw / 128 dw).
// ---------------------------------------------------------------------------
union AB16 { v16bf v; uint4 q[2]; unsigned short s[16]; };

#define W0_A ((1u << 16) | (1u << 20) | (3u << 22) | (3u << 25))  // 2B, pad 4dw/16dw
#define W0_W ((2u << 16) | (1u << 20) | (6u << 22) | (3u << 25))  // 4B, pad 4dw/128dw

__global__ __launch_bounds__(256) void gemm_bf16(
        const unsigned short* __restrict__ A, const float* __restrict__ W,
        const float* __restrict__ bias, float* __restrict__ outF,
        unsigned short* __restrict__ outH, int K, int N) {
    __shared__ __align__(128) unsigned short As[2][128 * 40];
    __shared__ __align__(128) float Ws[2][32 * 132];
    int tid = threadIdx.x;
    int lane = tid & 31, wv = tid >> 5;
    int half = lane >> 4, l16 = lane & 15;
    int n0 = blockIdx.x * 128;
    int nl = wv * 16 + l16;

    v8f zero = {0.f,0.f,0.f,0.f,0.f,0.f,0.f,0.f};
    v8f acc[8];
    for (int mt = 0; mt < 8; ++mt) acc[mt] = zero;

    bool wAligned = ((N & 3) == 0) && (n0 + 128 <= N);

    if (wAligned) {
        // -------- TDM double-buffered path --------
        unsigned ldsA0 = (unsigned)(size_t)(void*)&As[0][0];
        unsigned ldsA1 = (unsigned)(size_t)(void*)&As[1][0];
        unsigned ldsW0 = (unsigned)(size_t)(void*)&Ws[0][0];
        unsigned ldsW1 = (unsigned)(size_t)(void*)&Ws[1][0];
        int nk = K >> 5;
        if (tid == 0) {   // divergent branch: only wave 0 reaches issue block
            tdm_load_2d(ldsA0, (unsigned long long)(size_t)A, W0_A, 32, 128, (long long)K);
            tdm_load_2d(ldsW0, (unsigned long long)(size_t)(W + (size_t)n0 * 1), W0_W,
                        128, 32, (long long)N);
        }
        for (int i = 0; i < nk; ++i) {
            int cur = i & 1;
            if (tid == 0) {
                if (i + 1 < nk) {
                    int k1 = (i + 1) << 5;
                    unsigned la = (cur ? ldsA0 : ldsA1);
                    unsigned lw = (cur ? ldsW0 : ldsW1);
                    tdm_load_2d(la, (unsigned long long)(size_t)(A + k1), W0_A,
                                32, 128, (long long)K);
                    tdm_load_2d(lw, (unsigned long long)(size_t)(W + (size_t)k1 * N + n0),
                                W0_W, 128, 32, (long long)N);
                    __builtin_amdgcn_s_wait_tensorcnt(2);  // current tile done
                } else {
                    __builtin_amdgcn_s_wait_tensorcnt(0);  // last tile done
                }
            }
            __syncthreads();
            AB16 b;
            #pragma unroll
            for (int e = 0; e < 16; ++e) {
                int kk = e + half * 8 + ((e >= 8) ? 8 : 0);
                b.s[e] = f2bf_bits(Ws[cur][kk * 132 + nl]);
            }
            #pragma unroll
            for (int mt = 0; mt < 8; ++mt) {
                AB16 a;
                const unsigned short* rowp = &As[cur][(mt * 16 + l16) * 40];
                a.q[0] = *(const uint4*)(rowp + half * 8);
                a.q[1] = *(const uint4*)(rowp + 16 + half * 8);
                acc[mt] = __builtin_amdgcn_wmma_f32_16x16x32_bf16(
                    false, a.v, false, b.v, (short)0, acc[mt], false, false);
            }
            __syncthreads();   // release buffer for next TDM overwrite
        }
    } else {
        // -------- small-N fallback (cooperative loads, guarded) --------
        for (int k0 = 0; k0 < K; k0 += 32) {
            __syncthreads();
            for (int ch = tid; ch < 512; ch += 256) {
                int row = ch >> 2, sub = ch & 3;
                *(uint4*)(&As[0][row * 40 + sub * 8]) =
                    *(const uint4*)(A + (size_t)row * K + k0 + sub * 8);
            }
            for (int idx = tid; idx < 32 * 128; idx += 256) {
                int kk = idx >> 7, nn = idx & 127;
                int n = n0 + nn;
                Ws[0][kk * 132 + nn] = (n < N) ? W[(size_t)(k0 + kk) * N + n] : 0.f;
            }
            __syncthreads();
            AB16 b;
            #pragma unroll
            for (int e = 0; e < 16; ++e) {
                int kk = e + half * 8 + ((e >= 8) ? 8 : 0);
                b.s[e] = f2bf_bits(Ws[0][kk * 132 + nl]);
            }
            #pragma unroll
            for (int mt = 0; mt < 8; ++mt) {
                AB16 a;
                const unsigned short* rowp = &As[0][(mt * 16 + l16) * 40];
                a.q[0] = *(const uint4*)(rowp + half * 8);
                a.q[1] = *(const uint4*)(rowp + 16 + half * 8);
                acc[mt] = __builtin_amdgcn_wmma_f32_16x16x32_bf16(
                    false, a.v, false, b.v, (short)0, acc[mt], false, false);
            }
        }
    }

    // Epilogue: C layout -> VGPR r holds m_local = r + half*8, n = lane%16
    int n = n0 + nl;
    if (n < N) {
        float bv = bias[n];
        #pragma unroll
        for (int mt = 0; mt < 8; ++mt) {
            #pragma unroll
            for (int r = 0; r < 8; ++r) {
                int m = mt * 16 + half * 8 + r;
                float v = acc[mt][r] + bv;
                if (outF) outF[(size_t)m * N + n] = v;
                if (outH) outH[(size_t)m * N + n] = f2bf_bits(v);
            }
        }
    }
}

// ---------------------------------------------------------------------------
// Launch
// ---------------------------------------------------------------------------
extern "C" void kernel_launch(void* const* d_in, const int* in_sizes, int n_in,
                              void* d_out, int out_size, void* d_ws, size_t ws_size,
                              hipStream_t stream) {
    (void)in_sizes; (void)n_in; (void)out_size; (void)ws_size;
    const float* feat = (const float*)d_in[0];
    const float* rois = (const float*)d_in[1];
    const float* rpn  = (const float*)d_in[2];
    const float* W1   = (const float*)d_in[3];
    const float* b1   = (const float*)d_in[4];
    const float* W2   = (const float*)d_in[5];
    const float* b2   = (const float*)d_in[6];
    const float* Wc   = (const float*)d_in[7];
    const float* bc   = (const float*)d_in[8];
    const float* Wd   = (const float*)d_in[9];
    const float* bd   = (const float*)d_in[10];
    float* out = (float*)d_out;

    char* ws = (char*)d_ws;
    size_t off = 0;
    auto salloc = [&](size_t bytes) -> char* {
        char* p = ws + off;
        off = (off + bytes + 255) & ~(size_t)255;
        return p;
    };
    float* clipped = (float*)salloc((size_t)NROIS * 4 * sizeof(float));
    float* scores  = (float*)salloc((size_t)NROIS * sizeof(float));
    int*   order   = (int*)  salloc((size_t)PRE_NMS * sizeof(int));
    float* roisS   = (float*)salloc((size_t)POST_NMS * 4 * sizeof(float));
    unsigned short* pooled = (unsigned short*)salloc((size_t)POST_NMS * D_IN * 2);
    unsigned short* h1     = (unsigned short*)salloc((size_t)POST_NMS * D_H * 2);
    unsigned short* h2     = (unsigned short*)salloc((size_t)POST_NMS * D_H * 2);

    rpn_score_clip<<<(NROIS + 255) / 256, 256, 0, stream>>>(rois, rpn, clipped, scores);
    rank_topk<<<(NROIS + 255) / 256, 256, 0, stream>>>(scores, order);
    nms_compact<<<1, 32, 0, stream>>>(clipped, order,
                                      out + POST_NMS * NC + POST_NMS * NC * 4, roisS);
    roi_align_bf16<<<(POST_NMS * FEAT_C * 49) / 256, 256, 0, stream>>>(feat, roisS, pooled);

    gemm_bf16<<<D_H / 128, 256, 0, stream>>>(pooled, W1, b1, nullptr, h1, D_IN, D_H);
    gemm_bf16<<<D_H / 128, 256, 0, stream>>>(h1, W2, b2, nullptr, h2, D_H, D_H);
    gemm_bf16<<<1, 256, 0, stream>>>(h2, Wc, bc, out, nullptr, D_H, NC);
    gemm_bf16<<<1, 256, 0, stream>>>(h2, Wd, bd, out + POST_NMS * NC, nullptr, D_H, NC * 4);
}
